// GatingNetwork_5763846111396
// MI455X (gfx1250) — matrix operand order repaired and checked by reading, compiled, not run on hardware
//
#include <hip/hip_runtime.h>

typedef __attribute__((ext_vector_type(16))) __bf16 v16bf;
typedef __attribute__((ext_vector_type(8)))  float  v8f;

static __device__ __forceinline__ v8f wmma_bf16(v16bf a, v16bf b, v8f c) {
    // D = A(16x32 bf16) * B(32x16 bf16) + C(16x16 f32)
    return __builtin_amdgcn_wmma_f32_16x16x32_bf16(
        /*neg_a=*/false, a, /*neg_b=*/false, b,
        /*c_mod=*/(short)0, c, /*reuse_a=*/false, /*reuse_b=*/false);
}

#define WAVES 8
#define ROWS_PER_WAVE 16
#define ROWS_PER_BLOCK 128

__global__ __launch_bounds__(256) void gating_wmma_kernel(
    const float* __restrict__ e0, const float* __restrict__ e1, const float* __restrict__ e2,
    const unsigned char* __restrict__ mask,
    const float* __restrict__ W1,  const float* __restrict__ b1,
    const float* __restrict__ W2,  const float* __restrict__ b2,
    const float* __restrict__ Wc1, const float* __restrict__ bc1,
    const float* __restrict__ Wc2, const float* __restrict__ bc2,
    float* __restrict__ out_prob,  float* __restrict__ out_w,
    int B)
{
    // B-fragments for W1/Wc1, WMMA-ready: [ntile*2+kfrag][lane][elem]
    __shared__ alignas(32) __bf16 sW1f[16][32][16];
    __shared__ alignas(32) __bf16 sWcf[16][32][16];
    __shared__ float sB1[128], sW2[128], sBc1[128], sWc2[128];
    __shared__ float sB2, sBc2;
    __shared__ float sLogit[WAVES][ROWS_PER_WAVE][4];
    __shared__ float sWgt[WAVES][ROWS_PER_WAVE][4];

    const int tid = threadIdx.x;

    // ---- one-time weight staging: build B fragments in LDS (bf16) ----
    {
        const int frag  = tid >> 4;          // 0..15  (= ntile*2 + kfrag)
        const int kfrag = frag & 1;
        const int ntile = frag >> 1;
        const int l0    = tid & 15;
        #pragma unroll
        for (int half = 0; half < 2; ++half) {
            const int l  = l0 + half * 16;
            const int n  = ntile * 16 + (l & 15);
            const int kb = kfrag * 32 + ((l >= 16) ? 8 : 0);
            #pragma unroll
            for (int j = 0; j < 16; ++j) {
                const int k = kb + j + ((j >= 8) ? 8 : 0); // K index (0..63)
                sW1f[frag][l][j] = (__bf16)W1[k * 128 + n];
                sWcf[frag][l][j] = (__bf16)Wc1[k * 128 + n];
            }
        }
        if (tid < 128) {
            sB1[tid]  = b1[tid];  sW2[tid]  = W2[tid];
            sBc1[tid] = bc1[tid]; sWc2[tid] = Wc2[tid];
        }
        if (tid == 0) { sB2 = b2[0]; sBc2 = bc2[0]; }
    }
    __syncthreads();

    const int lane   = tid & 31;
    const int wv     = tid >> 5;
    const int mrow   = lane & 15;          // M within the 16-row tile
    const bool hiHalf = lane >= 16;

    const float* eptr[3] = { e0, e1, e2 };
    const int numTiles = B / ROWS_PER_BLOCK;

    for (int t = blockIdx.x; t < numTiles; t += gridDim.x) {
        const int row0 = t * ROWS_PER_BLOCK + wv * ROWS_PER_WAVE;
        const int m    = row0 + mrow;

        // ---- load A fragments (16x32 bf16) for 3 experts x 2 K-chunks ----
        // A layout: lane<16 -> elems 0..7:K=kb+0..7, elems 8..15:K=kb+16..23
        //           lane>=16 -> kb += 8
        v16bf a[3][2];
        #pragma unroll
        for (int e = 0; e < 3; ++e) {
            const float* rp = eptr[e] + (size_t)m * 64;
            #pragma unroll
            for (int kf = 0; kf < 2; ++kf) {
                const int kb = kf * 32 + (hiHalf ? 8 : 0);
                const float4* p0 = (const float4*)(rp + kb);
                const float4* p1 = (const float4*)(rp + kb + 16);
                float4 x0 = p0[0], x1 = p0[1], x2 = p1[0], x3 = p1[1];
                v16bf av;
                av[0]=(__bf16)x0.x;  av[1]=(__bf16)x0.y;  av[2]=(__bf16)x0.z;  av[3]=(__bf16)x0.w;
                av[4]=(__bf16)x1.x;  av[5]=(__bf16)x1.y;  av[6]=(__bf16)x1.z;  av[7]=(__bf16)x1.w;
                av[8]=(__bf16)x2.x;  av[9]=(__bf16)x2.y;  av[10]=(__bf16)x2.z; av[11]=(__bf16)x2.w;
                av[12]=(__bf16)x3.x; av[13]=(__bf16)x3.y; av[14]=(__bf16)x3.z; av[15]=(__bf16)x3.w;
                a[e][kf] = av;
            }
        }

        // ---- layer 1 per expert: h = relu(A*W1 + b1); logit = h . W2 + b2 ----
        #pragma unroll
        for (int e = 0; e < 3; ++e) {
            float part[8];
            #pragma unroll
            for (int i = 0; i < 8; ++i) part[i] = 0.f;

            #pragma unroll
            for (int nt = 0; nt < 8; ++nt) {
                v8f c = {};
                v16bf bA = *(const v16bf*)&sW1f[nt * 2 + 0][lane][0];
                v16bf bB = *(const v16bf*)&sW1f[nt * 2 + 1][lane][0];
                c = wmma_bf16(a[e][0], bA, c);
                c = wmma_bf16(a[e][1], bB, c);
                const int n = nt * 16 + mrow;
                const float bias = sB1[n], w2n = sW2[n];
                #pragma unroll
                for (int i = 0; i < 8; ++i) {
                    float h = c[i] + bias;
                    h = h > 0.f ? h : 0.f;
                    part[i] += h * w2n;
                }
            }
            // reduce over 16 lanes (columns) within each half-wave
            #pragma unroll
            for (int i = 0; i < 8; ++i) {
                #pragma unroll
                for (int d = 1; d < 16; d <<= 1)
                    part[i] += __shfl_xor(part[i], d, 32);
            }
            if (lane == 0) {
                #pragma unroll
                for (int i = 0; i < 8; ++i) sLogit[wv][i][e] = part[i] + sB2;
            }
            if (lane == 16) {
                #pragma unroll
                for (int i = 0; i < 8; ++i) sLogit[wv][8 + i][e] = part[i] + sB2;
            }
        }

        __builtin_amdgcn_wave_barrier();
        asm volatile("s_wait_dscnt 0x0" ::: "memory");

        // ---- masked softmax over 3 experts (lanes 0..15 each own one row) ----
        if (!hiHalf) {
            const int r = row0 + lane;
            const unsigned char m0 = mask[(size_t)r * 3 + 0];
            const unsigned char m1 = mask[(size_t)r * 3 + 1];
            const unsigned char m2 = mask[(size_t)r * 3 + 2];
            const float l0v = sLogit[wv][lane][0];
            const float l1v = sLogit[wv][lane][1];
            const float l2v = sLogit[wv][lane][2];
            const float NEG = -1e30f;
            const float a0 = m0 ? l0v : NEG;
            const float a1 = m1 ? l1v : NEG;
            const float a2 = m2 ? l2v : NEG;
            const float mx = fmaxf(a0, fmaxf(a1, a2));
            const float x0 = m0 ? __expf(a0 - mx) : 0.f;
            const float x1 = m1 ? __expf(a1 - mx) : 0.f;
            const float x2 = m2 ? __expf(a2 - mx) : 0.f;
            const float den = x0 + x1 + x2;
            const float inv = 1.f / fmaxf(den, 1e-30f);
            const bool any = (m0 | m1 | m2) != 0;
            const float w0 = any ? x0 * inv : 0.f;
            const float w1 = any ? x1 * inv : 0.f;
            const float w2 = any ? x2 * inv : 0.f;
            out_w[(size_t)r * 3 + 0] = w0;
            out_w[(size_t)r * 3 + 1] = w1;
            out_w[(size_t)r * 3 + 2] = w2;
            sWgt[wv][lane][0] = w0;
            sWgt[wv][lane][1] = w1;
            sWgt[wv][lane][2] = w2;
        }

        __builtin_amdgcn_wave_barrier();
        asm volatile("s_wait_dscnt 0x0" ::: "memory");

        // ---- fused = sum_e w_e * expert_e : per-lane scalar weights (A layout fixes M per lane)
        const float fw0 = sWgt[wv][mrow][0];
        const float fw1 = sWgt[wv][mrow][1];
        const float fw2 = sWgt[wv][mrow][2];
        v16bf f[2];
        #pragma unroll
        for (int kf = 0; kf < 2; ++kf) {
            #pragma unroll
            for (int j = 0; j < 16; ++j) {
                const float v = fw0 * (float)a[0][kf][j]
                              + fw1 * (float)a[1][kf][j]
                              + fw2 * (float)a[2][kf][j];
                f[kf][j] = (__bf16)v;
            }
        }

        // ---- layer 2: hc = relu(fused*Wc1 + bc1); prob = sigmoid(hc . Wc2 + bc2) ----
        float part2[8];
        #pragma unroll
        for (int i = 0; i < 8; ++i) part2[i] = 0.f;
        #pragma unroll
        for (int nt = 0; nt < 8; ++nt) {
            v8f c = {};
            v16bf bA = *(const v16bf*)&sWcf[nt * 2 + 0][lane][0];
            v16bf bB = *(const v16bf*)&sWcf[nt * 2 + 1][lane][0];
            c = wmma_bf16(f[0], bA, c);
            c = wmma_bf16(f[1], bB, c);
            const int n = nt * 16 + mrow;
            const float bias = sBc1[n], wc2n = sWc2[n];
            #pragma unroll
            for (int i = 0; i < 8; ++i) {
                float h = c[i] + bias;
                h = h > 0.f ? h : 0.f;
                part2[i] += h * wc2n;
            }
        }
        #pragma unroll
        for (int i = 0; i < 8; ++i) {
            #pragma unroll
            for (int d = 1; d < 16; d <<= 1)
                part2[i] += __shfl_xor(part2[i], d, 32);
        }
        if (lane == 0 || lane == 16) {
            const int rb = row0 + (hiHalf ? 8 : 0);
            #pragma unroll
            for (int i = 0; i < 8; ++i) {
                const float s = part2[i] + sBc2;
                out_prob[rb + i] = 1.f / (1.f + __expf(-s));
            }
        }
    }
}

extern "C" void kernel_launch(void* const* d_in, const int* in_sizes, int n_in,
                              void* d_out, int out_size, void* d_ws, size_t ws_size,
                              hipStream_t stream) {
    const float*         e0   = (const float*)d_in[0];
    const float*         e1   = (const float*)d_in[1];
    const float*         e2   = (const float*)d_in[2];
    const unsigned char* mask = (const unsigned char*)d_in[3]; // jnp.bool_ -> 1 byte
    const float*         W1   = (const float*)d_in[4];
    const float*         b1   = (const float*)d_in[5];
    const float*         W2   = (const float*)d_in[6];
    const float*         b2   = (const float*)d_in[7];
    const float*         Wc1  = (const float*)d_in[8];
    const float*         bc1  = (const float*)d_in[9];
    const float*         Wc2  = (const float*)d_in[10];
    const float*         bc2  = (const float*)d_in[11];

    const int B = in_sizes[0] / 64;          // e0 is (B, 64)
    float* out      = (float*)d_out;
    float* out_prob = out;                   // first B floats
    float* out_w    = out + B;               // then B*3 weights

    const int numTiles = B / ROWS_PER_BLOCK;
    int grid = numTiles < 1024 ? numTiles : 1024;
    if (grid < 1) grid = 1;

    gating_wmma_kernel<<<grid, 256, 0, stream>>>(
        e0, e1, e2, mask, W1, b1, W2, b2, Wc1, bc1, Wc2, bc2,
        out_prob, out_w, B);
}
